// NCDESelectiveSSMCell_29575144801152
// MI455X (gfx1250) — matrix operand order repaired and checked by reading, not compiled, not run on hardware
//
#include <hip/hip_runtime.h>
#include <hip/hip_bf16.h>
#include <math.h>

// NCDE selective SSM cell for gfx1250 (MI455X, wave32).
//
// delta GEMM (2048x2048x2080 incl. fused event@We.T tail): WMMA f16-in/f32-acc,
//   double-buffered LDS tiles filled with global_load_async_to_lds_b128 when the
//   toolchain exposes it (ASYNCcnt pipeline), else sync f16 loads + ds stores.
// state update: memory-bound streaming (~585 MB => ~25 us @ 23.3 TB/s), with
//   non-temporal hints on the touch-once h_prev/h_t traffic so the 192 MB L2
//   keeps the reused operands instead.

typedef __attribute__((ext_vector_type(16))) _Float16 v16h;
typedef __attribute__((ext_vector_type(8)))  _Float16 v8h;
typedef __attribute__((ext_vector_type(8)))  float    v8f;
typedef __attribute__((ext_vector_type(4)))  float    v4f;
typedef int gv4i __attribute__((vector_size(16)));   // matches builtin's V4i

#define D_MODEL 2048
#define D_STATE 16
#define D_EVENT 32
#define BATCH   2048
#define K_TOT   2080        // 2048 (x/Wd) + 32 (event/We) fused
#define K_TILES 65
#define LDS_STRIDE 40       // halves per row (32 data + 8 pad), keeps 16B align

#if __has_builtin(__builtin_amdgcn_global_load_async_to_lds_b128) && \
    __has_builtin(__builtin_amdgcn_s_wait_asynccnt)
#define HAVE_ASYNC_LDS 1
#else
#define HAVE_ASYNC_LDS 0
#endif

#if HAVE_ASYNC_LDS
// Prototype (from toolchain diagnostics): (v4i AS1*, v4i AS3*, imm offset, imm cpol)
#define ASYNC_CP16(gp, lp)                                                   \
  __builtin_amdgcn_global_load_async_to_lds_b128(                            \
      (__attribute__((address_space(1))) gv4i*)(gp),                         \
      (__attribute__((address_space(3))) gv4i*)(lp), 0, 0)
#endif

union Frag16 { v16h v; v8h h[2]; };

// ---------------------------------------------------------------------------
// Kernel 1: A_base[d,n] = -exp(clip(A_log[d,n], -3, 1))
// ---------------------------------------------------------------------------
__global__ __launch_bounds__(256) void prep_abase(const float* __restrict__ A_log,
                                                  float* __restrict__ Ab) {
  int i = blockIdx.x * 256 + threadIdx.x;
  if (i < D_MODEL * D_STATE) {
    float v = fminf(fmaxf(A_log[i], -3.0f), 1.0f);
    Ab[i] = -expf(v);
  }
}

// ---------------------------------------------------------------------------
// Kernel 1b: pack [main | tail] rows to f16 with K stride 2080.
//   dst[row, 0:2048)   = main[row, :]  (x or Wd)
//   dst[row, 2048:2080)= tail[row, :]  (event or We)
// ---------------------------------------------------------------------------
__global__ __launch_bounds__(256)
void pack_f16(const float* __restrict__ main_m, const float* __restrict__ tail_m,
              _Float16* __restrict__ dst) {
  int idx = blockIdx.x * 256 + threadIdx.x;        // segment of 16 halves
  if (idx >= 2048 * (K_TOT / 16)) return;
  int row = idx / (K_TOT / 16);
  int col = (idx % (K_TOT / 16)) * 16;
  const float4* s = (col < D_MODEL)
      ? (const float4*)(main_m + (size_t)row * D_MODEL + col)
      : (const float4*)(tail_m + (size_t)row * D_EVENT + (col - D_MODEL));
  float4 p0 = s[0], p1 = s[1], p2 = s[2], p3 = s[3];
  v8h h0 = {(_Float16)p0.x, (_Float16)p0.y, (_Float16)p0.z, (_Float16)p0.w,
            (_Float16)p1.x, (_Float16)p1.y, (_Float16)p1.z, (_Float16)p1.w};
  v8h h1 = {(_Float16)p2.x, (_Float16)p2.y, (_Float16)p2.z, (_Float16)p2.w,
            (_Float16)p3.x, (_Float16)p3.y, (_Float16)p3.z, (_Float16)p3.w};
  _Float16* d = dst + (size_t)row * K_TOT + col;
  *(v8h*)d = h0;
  *(v8h*)(d + 8) = h1;
}

// ---------------------------------------------------------------------------
// Shared GEMM pieces: fragment build + 8 WMMAs, and the softplus epilogue.
// ---------------------------------------------------------------------------
__device__ __forceinline__ void wmma_step(const _Float16* As, const _Float16* Bs,
                                          int mw, int nw, int l16, int ka, int kb,
                                          v8f acc[4][2]) {
  v16h afr[4], bfr[2];
  #pragma unroll
  for (int mt = 0; mt < 4; ++mt) {
    Frag16 ua;
    const _Float16* ap = &As[(mw + mt * 16 + l16) * LDS_STRIDE + ka];
    ua.h[0] = *(const v8h*)ap;          // K = ka+0..7
    ua.h[1] = *(const v8h*)(ap + 16);   // K = ka+16..23
    afr[mt] = ua.v;
  }
  #pragma unroll
  for (int nt = 0; nt < 2; ++nt) {
    Frag16 ub;
    const _Float16* bp = &Bs[(nw + nt * 16 + l16) * LDS_STRIDE + kb];
    ub.h[0] = *(const v8h*)bp;          // K = kb+0..7
    ub.h[1] = *(const v8h*)(bp + 16);   // K = kb+16..23
    bfr[nt] = ub.v;
  }
  #pragma unroll
  for (int mt = 0; mt < 4; ++mt)
    #pragma unroll
    for (int nt = 0; nt < 2; ++nt)
      acc[mt][nt] = __builtin_amdgcn_wmma_f32_16x16x32_f16(
          false, afr[mt], false, bfr[nt], (short)0, acc[mt][nt], false, false);
}

__device__ __forceinline__ void delta_epilogue(v8f acc[4][2], const float* bd,
                                               float* delta_out, int bM, int bN,
                                               int mw, int nw, int lane, int l16) {
  const int rbase = bM + mw + ((lane >= 16) ? 8 : 0);
  #pragma unroll
  for (int mt = 0; mt < 4; ++mt) {
    #pragma unroll
    for (int nt = 0; nt < 2; ++nt) {
      int col = bN + nw + nt * 16 + l16;
      float bias = bd[col];
      #pragma unroll
      for (int v = 0; v < 8; ++v) {
        int row = rbase + mt * 16 + v;
        float z = acc[mt][nt][v] + bias;
        float sp = (z > 20.0f) ? z : log1pf(expf(z));
        delta_out[(size_t)row * D_MODEL + col] = fminf(sp, 2.0f);
      }
    }
  }
}

// ---------------------------------------------------------------------------
// Kernel 2a: WMMA GEMM from packed f16 operands. Double-buffered LDS tiles,
// async fill (ASYNCcnt) when available.
// ---------------------------------------------------------------------------
__global__ __launch_bounds__(256)
void gemm_delta_packed(const _Float16* __restrict__ Xh, const _Float16* __restrict__ Wh,
                       const float* __restrict__ bd, float* __restrict__ delta_out) {
  __shared__ __align__(16) _Float16 SA[2][128 * LDS_STRIDE];
  __shared__ __align__(16) _Float16 SB[2][128 * LDS_STRIDE];

  const int tid  = threadIdx.x;
  const int lane = tid & 31;
  const int wv   = tid >> 5;
  const int mw   = (wv & 1) * 64;
  const int nw   = (wv >> 1) * 32;
  const int bM   = blockIdx.y * 128;   // batch base
  const int bN   = blockIdx.x * 128;   // d_model base
  const int l16  = lane & 15;
  const int ka   = (lane >= 16) ? 8 : 0;
  const int kb   = (lane >= 16) ? 8 : 0;

  v8f acc[4][2] = {};

#if HAVE_ASYNC_LDS
  // Each thread copies two 16B segments per matrix per tile (512 segs/tile).
  const int s0 = tid * 2;
  const int r0 = s0 >> 2, c0 = (s0 & 3) * 8;
  const int s1 = s0 + 1;
  const int r1 = s1 >> 2, c1 = (s1 & 3) * 8;

  #define ISSUE_TILE(kt, buf)                                                   \
    do {                                                                        \
      int k0 = (kt) * 32;                                                       \
      ASYNC_CP16(Xh + (size_t)(bM + r0) * K_TOT + k0 + c0,                      \
                 &SA[buf][r0 * LDS_STRIDE + c0]);                               \
      ASYNC_CP16(Xh + (size_t)(bM + r1) * K_TOT + k0 + c1,                      \
                 &SA[buf][r1 * LDS_STRIDE + c1]);                               \
      ASYNC_CP16(Wh + (size_t)(bN + r0) * K_TOT + k0 + c0,                      \
                 &SB[buf][r0 * LDS_STRIDE + c0]);                               \
      ASYNC_CP16(Wh + (size_t)(bN + r1) * K_TOT + k0 + c1,                      \
                 &SB[buf][r1 * LDS_STRIDE + c1]);                               \
    } while (0)

  ISSUE_TILE(0, 0);
  for (int kt = 0; kt < K_TILES; ++kt) {
    const int cur = kt & 1;
    if (kt + 1 < K_TILES) {
      ISSUE_TILE(kt + 1, cur ^ 1);              // prefetch next tile
      __builtin_amdgcn_s_wait_asynccnt(4);      // current tile's 4 ops done
    } else {
      __builtin_amdgcn_s_wait_asynccnt(0);
    }
    __syncthreads();                            // all waves' tile data visible
    wmma_step(SA[cur], SB[cur], mw, nw, l16, ka, kb, acc);
    __syncthreads();                            // reads done before buf reuse
  }
  #undef ISSUE_TILE
#else
  const int lr = tid >> 1;
  const int lc = (tid & 1) * 16;
  for (int kt = 0; kt < K_TILES; ++kt) {
    __syncthreads();
    const v8h* ga = (const v8h*)(Xh + (size_t)(bM + lr) * K_TOT + kt * 32 + lc);
    const v8h* gb = (const v8h*)(Wh + (size_t)(bN + lr) * K_TOT + kt * 32 + lc);
    v8h a0 = ga[0], a1 = ga[1], b0 = gb[0], b1 = gb[1];
    *(v8h*)&SA[0][lr * LDS_STRIDE + lc]     = a0;
    *(v8h*)&SA[0][lr * LDS_STRIDE + lc + 8] = a1;
    *(v8h*)&SB[0][lr * LDS_STRIDE + lc]     = b0;
    *(v8h*)&SB[0][lr * LDS_STRIDE + lc + 8] = b1;
    __syncthreads();
    wmma_step(SA[0], SB[0], mw, nw, l16, ka, kb, acc);
  }
#endif

  delta_epilogue(acc, bd, delta_out, bM, bN, mw, nw, lane, l16);
}

// ---------------------------------------------------------------------------
// Kernel 2b: fallback GEMM straight from f32 (convert on LDS store). Used only
// when workspace is too small for the packed f16 copies.
// ---------------------------------------------------------------------------
__global__ __launch_bounds__(256)
void gemm_delta_f32(const float* __restrict__ x, const float* __restrict__ ev,
                    const float* __restrict__ Wd, const float* __restrict__ We,
                    const float* __restrict__ bd, float* __restrict__ delta_out) {
  __shared__ __align__(16) _Float16 As[128 * LDS_STRIDE];
  __shared__ __align__(16) _Float16 Bs[128 * LDS_STRIDE];

  const int tid  = threadIdx.x;
  const int lane = tid & 31;
  const int wv   = tid >> 5;
  const int mw   = (wv & 1) * 64;
  const int nw   = (wv >> 1) * 32;
  const int bM   = blockIdx.y * 128;
  const int bN   = blockIdx.x * 128;
  const int lr   = tid >> 1;
  const int lc   = (tid & 1) * 16;
  const int l16  = lane & 15;
  const int ka   = (lane >= 16) ? 8 : 0;
  const int kb   = (lane >= 16) ? 8 : 0;

  v8f acc[4][2] = {};

  for (int kt = 0; kt < K_TILES; ++kt) {
    __syncthreads();
    float4 a0, a1, a2, a3, b0, b1, b2, b3;
    if (kt < 64) {
      const float4* ax = (const float4*)(x  + (size_t)(bM + lr) * D_MODEL + kt * 32 + lc);
      const float4* bw = (const float4*)(Wd + (size_t)(bN + lr) * D_MODEL + kt * 32 + lc);
      a0 = ax[0]; a1 = ax[1]; a2 = ax[2]; a3 = ax[3];
      b0 = bw[0]; b1 = bw[1]; b2 = bw[2]; b3 = bw[3];
    } else {
      const float4* ae = (const float4*)(ev + (size_t)(bM + lr) * D_EVENT + lc);
      const float4* be = (const float4*)(We + (size_t)(bN + lr) * D_EVENT + lc);
      a0 = ae[0]; a1 = ae[1]; a2 = ae[2]; a3 = ae[3];
      b0 = be[0]; b1 = be[1]; b2 = be[2]; b3 = be[3];
    }
    v8h ah0 = {(_Float16)a0.x, (_Float16)a0.y, (_Float16)a0.z, (_Float16)a0.w,
               (_Float16)a1.x, (_Float16)a1.y, (_Float16)a1.z, (_Float16)a1.w};
    v8h ah1 = {(_Float16)a2.x, (_Float16)a2.y, (_Float16)a2.z, (_Float16)a2.w,
               (_Float16)a3.x, (_Float16)a3.y, (_Float16)a3.z, (_Float16)a3.w};
    v8h bh0 = {(_Float16)b0.x, (_Float16)b0.y, (_Float16)b0.z, (_Float16)b0.w,
               (_Float16)b1.x, (_Float16)b1.y, (_Float16)b1.z, (_Float16)b1.w};
    v8h bh1 = {(_Float16)b2.x, (_Float16)b2.y, (_Float16)b2.z, (_Float16)b2.w,
               (_Float16)b3.x, (_Float16)b3.y, (_Float16)b3.z, (_Float16)b3.w};
    *(v8h*)&As[lr * LDS_STRIDE + lc]     = ah0;
    *(v8h*)&As[lr * LDS_STRIDE + lc + 8] = ah1;
    *(v8h*)&Bs[lr * LDS_STRIDE + lc]     = bh0;
    *(v8h*)&Bs[lr * LDS_STRIDE + lc + 8] = bh1;
    __syncthreads();
    wmma_step(As, Bs, mw, nw, l16, ka, kb, acc);
  }

  delta_epilogue(acc, bd, delta_out, bM, bN, mw, nw, lane, l16);
}

// ---------------------------------------------------------------------------
// Kernel 3: per batch-row small projections (N=16..32, not WMMA-shaped).
// ---------------------------------------------------------------------------
__global__ __launch_bounds__(64)
void row_small(const float* __restrict__ x, const float* __restrict__ ev,
               const float* __restrict__ B_W1, const float* __restrict__ B_b1,
               const float* __restrict__ C_W, const float* __restrict__ B_W2,
               const float* __restrict__ B_b2, const float* __restrict__ B_We,
               float* __restrict__ Bt, float* __restrict__ Ct) {
  const int b = blockIdx.x;
  const int t = threadIdx.x;
  __shared__ float U[32];

  float acc = 0.0f;
  if (t < 48) {
    const float4* xr = (const float4*)(x + (size_t)b * D_MODEL);
    const float4* wr = (t < 32) ? (const float4*)(B_W1 + (size_t)t * D_MODEL)
                                : (const float4*)(C_W + (size_t)(t - 32) * D_MODEL);
    for (int k = 0; k < D_MODEL / 4; ++k) {
      float4 a = xr[k], w = wr[k];
      acc += a.x * w.x + a.y * w.y + a.z * w.z + a.w * w.w;
    }
  }
  if (t < 32) U[t] = acc + B_b1[t];
  __syncthreads();

  if (t < 16) {
    float s = B_b2[t];
    #pragma unroll 4
    for (int j = 0; j < 32; ++j) {
      float u = U[j];
      s += (u / (1.0f + expf(-u))) * B_W2[t * 32 + j];  // silu(U) @ B_W2.T
    }
    #pragma unroll 4
    for (int e = 0; e < 32; ++e)
      s += ev[b * D_EVENT + e] * B_We[t * 32 + e];
    Bt[b * D_STATE + t] = s;
  }
  if (t >= 32 && t < 48) Ct[b * D_STATE + (t - 32)] = acc;
}

// ---------------------------------------------------------------------------
// Kernel 4: memory-bound state update, one thread per (b,d). Touch-once
// h_prev/h_t/delta/y traffic uses non-temporal hints (TH=NT) so L2 keeps the
// reused small tensors instead.
// ---------------------------------------------------------------------------
__global__ __launch_bounds__(256)
void state_update(const float* __restrict__ x, const float* __restrict__ hprev,
                  const float* __restrict__ ev, const float* __restrict__ A_We,
                  const float* __restrict__ Dv, const float* __restrict__ delta,
                  const float* __restrict__ Bt, const float* __restrict__ Ct,
                  const float* __restrict__ Ab, float* __restrict__ y_out,
                  float* __restrict__ h_out) {
  const int idx = blockIdx.x * 256 + threadIdx.x;   // 0 .. BATCH*D_MODEL-1
  const int b = idx >> 11;
  const int d = idx & (D_MODEL - 1);

  const float ds = __builtin_nontemporal_load(delta + idx);

  float ep = 0.0f;   // event pole: 32-length dot, rows L2-resident
  {
    const v4f* e4 = (const v4f*)(ev + (size_t)b * D_EVENT);
    const v4f* a4 = (const v4f*)(A_We + (size_t)d * D_EVENT);
    #pragma unroll
    for (int i = 0; i < D_EVENT / 4; ++i) {
      v4f e = e4[i], a = a4[i];
      ep += e.x * a.x + e.y * a.y + e.z * a.z + e.w * a.w;
    }
  }
  const float xv = x[idx];
  const v4f* hp4 = (const v4f*)(hprev + (size_t)idx * D_STATE);
  const v4f* ab4 = (const v4f*)(Ab + (size_t)d * D_STATE);
  const v4f* bt4 = (const v4f*)(Bt + (size_t)b * D_STATE);
  const v4f* ct4 = (const v4f*)(Ct + (size_t)b * D_STATE);
  v4f* ho4 = (v4f*)(h_out + (size_t)idx * D_STATE);

  float yacc = 0.0f;
  #pragma unroll
  for (int q = 0; q < D_STATE / 4; ++q) {
    v4f hp = __builtin_nontemporal_load(hp4 + q);
    v4f ab = ab4[q], bt = bt4[q], ct = ct4[q];
    v4f h;
    h.x = expf(ds * (ab.x + ep)) * hp.x + ds * bt.x * xv;
    h.y = expf(ds * (ab.y + ep)) * hp.y + ds * bt.y * xv;
    h.z = expf(ds * (ab.z + ep)) * hp.z + ds * bt.z * xv;
    h.w = expf(ds * (ab.w + ep)) * hp.w + ds * bt.w * xv;
    yacc += h.x * ct.x + h.y * ct.y + h.z * ct.z + h.w * ct.w;
    __builtin_nontemporal_store(h, ho4 + q);
  }
  __builtin_nontemporal_store(yacc + Dv[d] * xv, y_out + idx);
}

// ---------------------------------------------------------------------------
extern "C" void kernel_launch(void* const* d_in, const int* in_sizes, int n_in,
                              void* d_out, int out_size, void* d_ws, size_t ws_size,
                              hipStream_t stream) {
  const float* x     = (const float*)d_in[0];
  const float* hprev = (const float*)d_in[1];
  const float* ev    = (const float*)d_in[2];
  const float* A_log = (const float*)d_in[3];
  const float* Dv    = (const float*)d_in[4];
  const float* Wd    = (const float*)d_in[5];
  const float* bd    = (const float*)d_in[6];
  const float* We    = (const float*)d_in[7];
  const float* B_W1  = (const float*)d_in[8];
  const float* B_b1  = (const float*)d_in[9];
  const float* B_W2  = (const float*)d_in[10];
  const float* B_b2  = (const float*)d_in[11];
  const float* B_We  = (const float*)d_in[12];
  const float* A_We  = (const float*)d_in[13];
  const float* C_W   = (const float*)d_in[14];

  float* ws    = (float*)d_ws;
  float* delta = ws;                                   // BATCH*D_MODEL f32
  float* Bt    = delta + (size_t)BATCH * D_MODEL;      // BATCH*D_STATE
  float* Ct    = Bt + (size_t)BATCH * D_STATE;
  float* Ab    = Ct + (size_t)BATCH * D_STATE;
  float* f32_end = Ab + (size_t)D_MODEL * D_STATE;

  const size_t f32_bytes = (size_t)((char*)f32_end - (char*)d_ws);
  const size_t pack_bytes = (size_t)2048 * K_TOT * sizeof(_Float16);  // per matrix
  const bool use_packed = ws_size >= f32_bytes + 2 * pack_bytes;

  float* y_out = (float*)d_out;
  float* h_out = y_out + (size_t)BATCH * D_MODEL;

  prep_abase<<<(D_MODEL * D_STATE + 255) / 256, 256, 0, stream>>>(A_log, Ab);

  dim3 g(D_MODEL / 128, BATCH / 128);
  if (use_packed) {
    _Float16* Xh = (_Float16*)((char*)d_ws + f32_bytes);
    _Float16* Wh = Xh + (size_t)2048 * K_TOT;
    int segs = 2048 * (K_TOT / 16);
    pack_f16<<<(segs + 255) / 256, 256, 0, stream>>>(x, ev, Xh);
    pack_f16<<<(segs + 255) / 256, 256, 0, stream>>>(Wd, We, Wh);
    gemm_delta_packed<<<g, 256, 0, stream>>>(Xh, Wh, bd, delta);
  } else {
    gemm_delta_f32<<<g, 256, 0, stream>>>(x, ev, Wd, We, bd, delta);
  }

  row_small<<<BATCH, 64, 0, stream>>>(x, ev, B_W1, B_b1, C_W, B_W2, B_b2, B_We, Bt, Ct);

  state_update<<<(BATCH * D_MODEL) / 256, 256, 0, stream>>>(
      x, hprev, ev, A_We, Dv, delta, Bt, Ct, Ab, y_out, h_out);
}
